// SVS_38457137168866
// MI455X (gfx1250) — matrix operand order repaired and checked by reading, not compiled, not run on hardware
//
#include <hip/hip_runtime.h>
#include <math.h>

// ---------------------------------------------------------------------------
// Problem constants (match reference)
// ---------------------------------------------------------------------------
#define Hn 256
#define Wn 512
#define Bn 32
#define HW (Hn * Wn)
#define PI_F 3.14159265358979323846f
#define DEPTH_TH 10.0f
#define BASELINE_F 0.26f
#define ALPHA_F 0.85f
#define LAMBDA_NORM_F 0.05f
#define MMAP_COS_TH 0.9f
#define EPS_F 1e-6f

typedef __attribute__((ext_vector_type(16))) _Float16 v16h;
typedef __attribute__((ext_vector_type(8)))  float    v8f;

// ---------------------------------------------------------------------------
// Small device helpers
// ---------------------------------------------------------------------------
__device__ __forceinline__ float thetaOf(int v) {
    return (((float)v + 0.5f) / (float)Hn - 0.5f) * PI_F;
}
__device__ __forceinline__ float phiOf(int u) {
    return (((float)u + 0.5f) / (float)Wn - 0.5f) * 2.0f * PI_F;
}
__device__ __forceinline__ float attnOf(int v) {
    float a = 1.0f - fabsf(thetaOf(v)) / (PI_F * 0.5f);
    return fminf(fmaxf(a, 0.0f), 1.0f);
}
__device__ __forceinline__ void xyzOf(const float* __restrict__ depth,
                                      int b, int v, int u,
                                      float* ox, float* oy, float* oz) {
    float d  = depth[(size_t)b * HW + (size_t)v * Wn + u];
    float th = thetaOf(v), ph = phiOf(u);
    float ct = cosf(th), st = sinf(th), cp = cosf(ph), sp = sinf(ph);
    *ox = d * ct * cp; *oy = d * st; *oz = d * ct * sp;
}
__device__ __forceinline__ void gauss7(float* g) {
    float s = 0.0f;
#pragma unroll
    for (int i = 0; i < 7; ++i) {
        float ax = (float)(i - 3);
        g[i] = __expf(-ax * ax / 4.5f);
        s += g[i];
    }
#pragma unroll
    for (int i = 0; i < 7; ++i) g[i] /= s;
}

__device__ __forceinline__ v8f wmma16(v16h a, v16h b, v8f c) {
    // D = A(16x32,f16) * B(32x16,f16) + C(16x16,f32)  -> v_wmma_f32_16x16x32_f16
    return __builtin_amdgcn_wmma_f32_16x16x32_f16(
        /*neg_a=*/false, a, /*neg_b=*/false, b,
        /*c_mod=*/(short)0, c, /*reuse_a=*/false, /*reuse_b=*/false);
}

// lane <-> lane^16 exchange via ds_swizzle SWAPX16 (ISA §11: xor=0x10, and=0x1f)
__device__ __forceinline__ int lane_swap16(int x) {
    return __builtin_amdgcn_ds_swizzle(x, 0x401f);
}
// full-wave butterfly sum via ds_swizzle (all lanes end with the total)
__device__ __forceinline__ float wave_sum(float x) {
    x += __int_as_float(__builtin_amdgcn_ds_swizzle(__float_as_int(x), 0x401f)); // xor 16
    x += __int_as_float(__builtin_amdgcn_ds_swizzle(__float_as_int(x), 0x201f)); // xor 8
    x += __int_as_float(__builtin_amdgcn_ds_swizzle(__float_as_int(x), 0x101f)); // xor 4
    x += __int_as_float(__builtin_amdgcn_ds_swizzle(__float_as_int(x), 0x081f)); // xor 2
    x += __int_as_float(__builtin_amdgcn_ds_swizzle(__float_as_int(x), 0x041f)); // xor 1
    return x;
}

// 2-value block reduction, result valid on tid==0
__device__ __forceinline__ void blockReduce2(float& a, float& b, float* sbuf) {
    a = wave_sum(a);
    b = wave_sum(b);
    int lane = threadIdx.x & 31, wave = threadIdx.x >> 5;
    int nw = blockDim.x >> 5;
    if (lane == 0) { sbuf[wave] = a; sbuf[16 + wave] = b; }
    __syncthreads();
    if (threadIdx.x == 0) {
        float sa = 0.0f, sb = 0.0f;
        for (int w = 0; w < nw; ++w) { sa += sbuf[w]; sb += sbuf[16 + w]; }
        a = sa; b = sb;
    }
}

// ---------------------------------------------------------------------------
// K0: zero workspace (scalars + splat accumulators) -- graph-replay safe
// ---------------------------------------------------------------------------
__global__ void k_zero(float* __restrict__ p, size_t n) {
    size_t i = (size_t)blockIdx.x * blockDim.x + threadIdx.x;
    size_t stride = (size_t)gridDim.x * blockDim.x;
    for (; i < n; i += stride) p[i] = 0.0f;
}

// ---------------------------------------------------------------------------
// K1: forward bilinear splat of left_rgb along the vertical disparity.
// cu == u exactly; cv = v + dtheta_pix. fp32 atomics into planar accumulators.
// ---------------------------------------------------------------------------
__global__ __launch_bounds__(256) void k_splat(
    const float* __restrict__ rgb, const float* __restrict__ depth,
    float* __restrict__ accW, float* __restrict__ accR,
    float* __restrict__ accG, float* __restrict__ accB) {
    int i = blockIdx.x * 256 + threadIdx.x;
    if (i >= Bn * HW) return;
    int b = i / HW, rem = i - b * HW;
    int v = rem / Wn, u = rem - v * Wn;

    float d   = depth[i];
    float th  = thetaOf(v);
    float th2 = atan2f(d * sinf(th) - BASELINE_F, d * cosf(th));
    float cu  = (float)u;
    float cv  = (float)v + (th2 - th) * ((float)Hn / PI_F);
    if (!(fabsf(cu) < 1e30f)) cu = 0.0f;   // non-finite -> 0 (matches reference)
    if (!(fabsf(cv) < 1e30f)) cv = 0.0f;
    bool valid = d < DEPTH_TH;

    float u0 = floorf(cu), v0 = floorf(cv);
    float du = cu - u0,    dv = cv - v0;
    float r  = rgb[(size_t)(b * 3 + 0) * HW + rem];
    float g  = rgb[(size_t)(b * 3 + 1) * HW + rem];
    float bl = rgb[(size_t)(b * 3 + 2) * HW + rem];

    float cw[4]  = { (1 - du) * (1 - dv), du * (1 - dv), (1 - du) * dv, du * dv };
    float cuu[4] = { u0, u0 + 1, u0, u0 + 1 };
    float cvv[4] = { v0, v0, v0 + 1, v0 + 1 };
#pragma unroll
    for (int c = 0; c < 4; ++c) {
        bool inb = (cuu[c] >= 0.0f) && (cuu[c] < (float)Wn) &&
                   (cvv[c] >= 0.0f) && (cvv[c] < (float)Hn) && valid;
        float w = inb ? cw[c] : 0.0f;
        if (w != 0.0f) {
            int ui = min(max((int)cuu[c], 0), Wn - 1);
            int vi = min(max((int)cvv[c], 0), Hn - 1);
            size_t j = (size_t)b * HW + (size_t)vi * Wn + ui;
            atomicAdd(&accW[j], w);
            atomicAdd(&accR[j], r * w);
            atomicAdd(&accG[j], g * w);
            atomicAdd(&accB[j], bl * w);
        }
    }
}

// ---------------------------------------------------------------------------
// K2: fused SSIM + L1 photometric loss. One block = one 16x16 output tile.
// The 7x7 Gaussian is separable; each 7-tap pass over a 16-wide tile is a
// GEMM against a constant banded matrix -> v_wmma_f32_16x16x32_f16.
//   H(16x16) = Patch(16x32) x Bband(32x16)   (2x for the 32 patch rows)
//   D(16x16) = Aband(16x32) x H(32x16)
// Layouts per CDNA5 ISA 7.12.2:
//   A 16x32 f16 : lane&15 = row M; K per-elem e -> e + (e>=8?8:0) + (lane>=16?8:0)
//   B 32x16 f16 : lane&15 = col N; K map as for A; pass 2 uses K = e (lane<16) or e+16 (lane>=16) in both vA and B
//   D 16x16 f32 : lane&15 = N;     M = e + (lane>=16?8:0)
// D->B conversion between passes: pack to f16 pairs first, then a single
// ds_swizzle SWAPX16 per 32-bit pair (8 DS ops instead of 16 bpermutes).
// ---------------------------------------------------------------------------
__global__ __launch_bounds__(128) void k_photo(
    const float* __restrict__ up_rgb, const float* __restrict__ up_depth,
    const float* __restrict__ accW,  const float* __restrict__ accR,
    const float* __restrict__ accG,  const float* __restrict__ accB,
    float* __restrict__ scal) {
    __shared__ float sX[3][32][33];     // warped image patch (halo 3, band pad)
    __shared__ float sY[3][32][33];     // target image patch
    __shared__ float sConv[15][16][17]; // 3 ch x {mx,my,E[xx],E[yy],E[xy]}
    __shared__ float sRed[32];

    const int b = blockIdx.z;
    const int R = blockIdx.y * 16;
    const int C = blockIdx.x * 16;
    const int tid = threadIdx.x;
    const int lane = tid & 31;
    const int wave = tid >> 5;

    // -------- load 32x32 patches (zero 'SAME' padding) --------
    for (int t = tid; t < 3 * 32 * 32; t += 128) {
        int ch = t >> 10, p = t & 1023, pr = p >> 5, pc = p & 31;
        int gv = R - 3 + pr, gu = C - 3 + pc;
        float xv = 0.0f, yv = 0.0f;
        if (gv >= 0 && gv < Hn && gu >= 0 && gu < Wn) {
            size_t pix = (size_t)b * HW + (size_t)gv * Wn + gu;
            const float* ap = (ch == 0) ? accR : ((ch == 1) ? accG : accB);
            xv = ap[pix] / fmaxf(accW[pix], EPS_F);
            yv = up_rgb[(size_t)(b * 3 + ch) * HW + (size_t)gv * Wn + gu];
        }
        sX[ch][pr][pc] = xv;
        sY[ch][pr][pc] = yv;
    }
    __syncthreads();

    float g[7]; gauss7(g);

    // -------- constant banded fragments (same for all 15 conv jobs) --------
    v16h hB, vA;
    {
        int n = lane & 15;
#pragma unroll
        for (int e = 0; e < 16; ++e) {
            int kB = e + ((e >= 8) ? 8 : 0) + ((lane >= 16) ? 8 : 0);
            int kA = (lane < 16) ? e : e + 16;
            int dB = kB - n;  // horiz: weight g[k-n], band k-n in [0,6]
            int dA = kA - n;  // vert:  weight g[k-m], m == lane&15 == n
            hB[e] = (_Float16)((dB >= 0 && dB <= 6) ? g[dB] : 0.0f);
            vA[e] = (_Float16)((dA >= 0 && dA <= 6) ? g[dA] : 0.0f);
        }
    }

    // -------- 15 separable convolutions, 4 waves round-robin --------
    for (int pair = wave; pair < 15; pair += 4) {
        int ch = pair / 5, q = pair % 5;
        v16h a0, a1;
        int m = lane & 15;
#pragma unroll
        for (int e = 0; e < 16; ++e) {
            int k = e + ((e >= 8) ? 8 : 0) + ((lane >= 16) ? 8 : 0);
            float x0 = sX[ch][m][k],      y0 = sY[ch][m][k];
            float x1 = sX[ch][m + 16][k], y1 = sY[ch][m + 16][k];
            float v0, v1;
            switch (q) {
                case 0:  v0 = x0;      v1 = x1;      break;
                case 1:  v0 = y0;      v1 = y1;      break;
                case 2:  v0 = x0 * x0; v1 = x1 * x1; break;
                case 3:  v0 = y0 * y0; v1 = y1 * y1; break;
                default: v0 = x0 * y0; v1 = x1 * y1; break;
            }
            a0[e] = (_Float16)v0;
            a1[e] = (_Float16)v1;
        }
        v8f z = {};
        v8f h0 = wmma16(a0, hB, z);   // hconv rows 0..15 of patch
        v8f h1 = wmma16(a1, hB, z);   // hconv rows 16..31

        // Pack to f16 pairs, then exchange lane<->lane^16 per packed dword.
        union HF { v16h h; int i[8]; } hf, bff;
#pragma unroll
        for (int e = 0; e < 8; ++e) {
            hf.h[e]     = (_Float16)h0[e];   // i[0..3]: h0 pairs
            hf.h[e + 8] = (_Float16)h1[e];   // i[4..7]: h1 pairs
        }
#pragma unroll
        for (int k = 0; k < 4; ++k) {
            int own0 = hf.i[k];
            int own1 = hf.i[k + 4];
            int oth0 = lane_swap16(own0);
            int oth1 = lane_swap16(own1);
            // B-operand: lanes<16 keep Hc K=0..15 (own h0 + partner h0),
            //            lanes>=16 keep K=16..31 (partner h1 + own h1)
            bff.i[k]     = (lane < 16) ? own0 : oth1;
            bff.i[k + 4] = (lane < 16) ? oth0 : own1;
        }
        v8f dres = wmma16(vA, bff.h, z);
#pragma unroll
        for (int e = 0; e < 8; ++e) {
            int mm = e + ((lane >= 16) ? 8 : 0);
            sConv[pair][mm][lane & 15] = dres[e];
        }
    }
    __syncthreads();

    // -------- per-pixel SSIM + L1 + attention-weighted accumulation --------
    const float C1 = 1e-4f, C2 = 9e-4f;
    float s_plwm = 0.0f, s_wm = 0.0f;
    for (int p = tid; p < 256; p += 128) {
        int m = p >> 4, n = p & 15;
        int v = R + m, u = C + n;
        float dssim = 0.0f, l1 = 0.0f;
#pragma unroll
        for (int ch = 0; ch < 3; ++ch) {
            float mx  = sConv[ch * 5 + 0][m][n];
            float my  = sConv[ch * 5 + 1][m][n];
            float sxx = sConv[ch * 5 + 2][m][n] - mx * mx;
            float syy = sConv[ch * 5 + 3][m][n] - my * my;
            float sxy = sConv[ch * 5 + 4][m][n] - mx * my;
            float ssim = (2.0f * mx * my + C1) * (2.0f * sxy + C2) /
                         ((mx * mx + my * my + C1) * (sxx + syy + C2));
            dssim += (1.0f - ssim) * 0.5f;
            l1 += fabsf(sX[ch][m + 3][n + 3] - sY[ch][m + 3][n + 3]);
        }
        float pl  = ALPHA_F * (dssim * (1.0f / 3.0f)) +
                    (1.0f - ALPHA_F) * (l1 * (1.0f / 3.0f));
        float cut = (up_depth[(size_t)b * HW + (size_t)v * Wn + u] < DEPTH_TH) ? 1.0f : 0.0f;
        float wm  = cut * attnOf(v);
        s_plwm += pl * wm;
        s_wm   += wm;
    }
    blockReduce2(s_plwm, s_wm, sRed);
    if (tid == 0) {
        atomicAdd(&scal[0], s_plwm);
        atomicAdd(&scal[1], s_wm);
    }
}

// ---------------------------------------------------------------------------
// K3: edge-aware smoothness loss on back-projected XYZ
// ---------------------------------------------------------------------------
__global__ __launch_bounds__(256) void k_smooth(
    const float* __restrict__ rgb, const float* __restrict__ depth,
    const float* __restrict__ up_depth, float* __restrict__ scal) {
    __shared__ float sRed[32];
    int i = blockIdx.x * 256 + threadIdx.x;
    int b = i / HW, rem = i - b * HW;
    int v = rem / Wn, u = rem - v * Wn;

    float px, py, pz;
    xyzOf(depth, b, v, u, &px, &py, &pz);

    float su = 0.0f, gu = 0.0f;
    if (u < Wn - 1) {
        float qx, qy, qz;
        xyzOf(depth, b, v, u + 1, &qx, &qy, &qz);
        su = fabsf(qx - px) + fabsf(qy - py) + fabsf(qz - pz);
#pragma unroll
        for (int c = 0; c < 3; ++c) {
            size_t base = (size_t)(b * 3 + c) * HW + (size_t)v * Wn + u;
            gu += fabsf(rgb[base + 1] - rgb[base]);
        }
    }
    float sv = 0.0f, gvv = 0.0f;
    if (v < Hn - 1) {
        float qx, qy, qz;
        xyzOf(depth, b, v + 1, u, &qx, &qy, &qz);
        sv = fabsf(qx - px) + fabsf(qy - py) + fabsf(qz - pz);
#pragma unroll
        for (int c = 0; c < 3; ++c) {
            size_t base = (size_t)(b * 3 + c) * HW + (size_t)v * Wn + u;
            gvv += fabsf(rgb[base + Wn] - rgb[base]);
        }
    }
    float cut  = (up_depth[i] < DEPTH_TH) ? 1.0f : 0.0f;
    float wsm  = (1.0f - attnOf(v)) * cut;
    float term = (su * __expf(-gu) + sv * __expf(-gvv)) * wsm;

    blockReduce2(term, cut, sRed);
    if (threadIdx.x == 0) {
        atomicAdd(&scal[2], term);
        atomicAdd(&scal[3], cut);
    }
}

// ---------------------------------------------------------------------------
// K4: vanishing-point normal loss (rolled cross-product normals, argmax of 6)
// ---------------------------------------------------------------------------
__global__ __launch_bounds__(256) void k_norm(
    const float* __restrict__ depth, const float* __restrict__ vps,
    float* __restrict__ scal) {
    __shared__ float sRed[32];
    int i = blockIdx.x * 256 + threadIdx.x;
    int b = i / HW, rem = i - b * HW;
    int v = rem / Wn, u = rem - v * Wn;

    // xyz at rolled positions (jnp.roll wraps)
    float ax, ay, az, bx, by, bz, cx, cy, cz, dx, dy, dz;
    xyzOf(depth, b, v, (u + 3) & (Wn - 1), &ax, &ay, &az);
    xyzOf(depth, b, v, (u - 3 + Wn) & (Wn - 1), &bx, &by, &bz);
    xyzOf(depth, b, (v + 3) & (Hn - 1), u, &cx, &cy, &cz);
    xyzOf(depth, b, (v - 3 + Hn) & (Hn - 1), u, &dx, &dy, &dz);
    float dux = ax - bx, duy = ay - by, duz = az - bz;  // dpu
    float dvx = cx - dx, dvy = cy - dy, dvz = cz - dz;  // dpv
    // n = cross(dpv, dpu)
    float nx = dvy * duz - dvz * duy;
    float ny = dvz * dux - dvx * duz;
    float nz = dvx * duy - dvy * dux;
    float len = sqrtf(nx * nx + ny * ny + nz * nz);
    float inv = 1.0f / (len + EPS_F);
    nx *= inv; ny *= inv; nz *= inv;

    // 6 candidates: +/- normalized vps rows
    float cand[6][3];
#pragma unroll
    for (int k = 0; k < 3; ++k) {
        float vx = vps[(size_t)b * 9 + k * 3 + 0];
        float vy = vps[(size_t)b * 9 + k * 3 + 1];
        float vz = vps[(size_t)b * 9 + k * 3 + 2];
        float vl = sqrtf(vx * vx + vy * vy + vz * vz);
        float vi = 1.0f / (vl + EPS_F);
        cand[k][0] = vx * vi;  cand[k][1] = vy * vi;  cand[k][2] = vz * vi;
        cand[k + 3][0] = -cand[k][0];
        cand[k + 3][1] = -cand[k][1];
        cand[k + 3][2] = -cand[k][2];
    }
    float best = -1e30f; int bi = 0;
#pragma unroll
    for (int k = 0; k < 6; ++k) {
        float s = cand[k][0] * nx + cand[k][1] * ny + cand[k][2] * nz;
        if (s > best) { best = s; bi = k; }
    }
    float mask = (best > MMAP_COS_TH) ? 1.0f : 0.0f;
    float alx = cand[bi][0], aly = cand[bi][1], alz = cand[bi][2];
    float nl = sqrtf(nx * nx + ny * ny + nz * nz);
    float al = sqrtf(alx * alx + aly * aly + alz * alz);
    float cs = (nx * alx + ny * aly + nz * alz) / (nl * al + EPS_F);
    if (cs != cs) cs = 1.0f;  // NaN -> 1
    float contrib = (1.0f - cs) * mask;

    float dummy = 0.0f;
    blockReduce2(contrib, dummy, sRed);
    if (threadIdx.x == 0) atomicAdd(&scal[4], contrib);
}

// ---------------------------------------------------------------------------
// K5: combine scalars
// ---------------------------------------------------------------------------
__global__ void k_final(const float* __restrict__ scal, float* __restrict__ out) {
    if (blockIdx.x == 0 && threadIdx.x == 0) {
        float photo  = scal[0] / (scal[1] + EPS_F);
        float smooth = scal[2] / (scal[3] + EPS_F);
        float nrm    = scal[4] / (float)((size_t)Bn * HW);
        out[0] = photo + 0.1f * smooth + LAMBDA_NORM_F * nrm;
    }
}

// ---------------------------------------------------------------------------
// Launch
// ---------------------------------------------------------------------------
extern "C" void kernel_launch(void* const* d_in, const int* in_sizes, int n_in,
                              void* d_out, int out_size, void* d_ws, size_t ws_size,
                              hipStream_t stream) {
    const float* left_rgb   = (const float*)d_in[0];
    const float* left_depth = (const float*)d_in[1];
    const float* up_depth   = (const float*)d_in[2];
    const float* up_rgb     = (const float*)d_in[3];
    const float* vps        = (const float*)d_in[4];
    // d_in[5] = epoch (unused by the loss)

    const size_t N = (size_t)Bn * HW;  // 4,194,304
    float* ws   = (float*)d_ws;
    float* scal = ws;            // 64 scalar slots
    float* accW = ws + 64;       // N
    float* accR = accW + N;      // N
    float* accG = accR + N;      // N
    float* accB = accG + N;      // N  (total: 64 MiB + 256 B)

    k_zero<<<4096, 256, 0, stream>>>(ws, 64 + 4 * N);
    k_splat<<<(int)(N / 256), 256, 0, stream>>>(left_rgb, left_depth,
                                                accW, accR, accG, accB);
    dim3 g2(Wn / 16, Hn / 16, Bn);  // 32 x 16 x 32 = 16384 tiles
    k_photo<<<g2, 128, 0, stream>>>(up_rgb, up_depth, accW, accR, accG, accB, scal);
    k_smooth<<<(int)(N / 256), 256, 0, stream>>>(left_rgb, left_depth, up_depth, scal);
    k_norm<<<(int)(N / 256), 256, 0, stream>>>(left_depth, vps, scal);
    k_final<<<1, 64, 0, stream>>>(scal, (float*)d_out);
}
